// SecondaryTimeCapsA_51934744543555
// MI455X (gfx1250) — compile-verified, matrix operand order checked by reading, not run on hardware
//
#include <hip/hip_runtime.h>

typedef _Float16 v16h __attribute__((ext_vector_type(16)));
typedef float    v8f  __attribute__((ext_vector_type(8)));

#define B_SZ 8
#define T_SZ 1024
#define NP 32
#define DP 8
#define NT 16
#define DT 16
#define O_SZ 256
#define TPB 16            // t-values per block
#define THREADS 128       // 4 waves
#define WAVES 4
#define T_PER_WAVE 4

union FragU { v16h v; unsigned u[8]; };

__device__ __forceinline__ unsigned pk2(float a, float b) {
  unsigned short lo = __builtin_bit_cast(unsigned short, (_Float16)a);
  unsigned short hi = __builtin_bit_cast(unsigned short, (_Float16)b);
  return (unsigned)lo | ((unsigned)hi << 16);
}
__device__ __forceinline__ _Float16 h16(unsigned w, int h) {
  return __builtin_bit_cast(_Float16, (unsigned short)(w >> (16 * h)));
}

__global__ __launch_bounds__(THREADS) void caps_wmma_kernel(
    const float* __restrict__ x, const float* __restrict__ ck,
    const float* __restrict__ cb, const float* __restrict__ bw,
    float* __restrict__ out)
{
  // ---- LDS (total ~152 KB) ----
  __shared__ unsigned       kfH[16 * 32 * 8];           // 16 KB  K-frags hi (pre-swizzled dwords)
  __shared__ unsigned       kfL[16 * 32 * 8];           // 16 KB  K-frags lo
  __shared__ unsigned short xsH[(TPB + 2) * NP * DP];   //  9 KB  x slab hi (f16 bits)
  __shared__ unsigned short xsL[(TPB + 2) * NP * DP];   //  9 KB  x slab lo
  __shared__ float          bws[TPB * NT * NP];         // 32 KB  B_weight slab
  __shared__ unsigned       ybuf[WAVES][O_SZ * 16];     // 64 KB  Y staging: [o][p-pair] f16x2
  __shared__ unsigned short vbuf[WAVES][O_SZ];          //  2 KB  v (f16 bits)
  __shared__ unsigned short wbuf[WAVES][NT * NP];       //  4 KB  w = softmax+Bw (f16 bits)

  const int tid  = threadIdx.x;
  const int lane = tid & 31;
  const int wave = tid >> 5;
  const int col  = lane & 15;
  const bool hiL = lane >= 16;

  const int b  = blockIdx.x >> 6;
  const int t0 = (blockIdx.x & 63) * TPB;

  // ================= Phase A: fill LDS =================
  // Conv kernel fragments, bias folded at k==24, K padded 24->32 with zeros.
  for (int i = tid; i < 16 * 32 * 8; i += THREADS) {
    int r  = i & 7;
    int ln = (i >> 3) & 31;
    int m  = i >> 8;
    int o  = m * 16 + (ln & 15);
    int kb = 2 * (r & 3) + ((r >= 4) ? 16 : 0) + ((ln >= 16) ? 8 : 0);
    unsigned hw = 0u, lw = 0u;
    for (int h = 0; h < 2; ++h) {
      int k = kb + h;
      float f = 0.0f;
      if (k < 24)       f = ck[k * O_SZ + o];
      else if (k == 24) f = cb[o];
      _Float16 fh = (_Float16)f;
      _Float16 fl = (_Float16)(f - (float)fh);
      hw |= (unsigned)__builtin_bit_cast(unsigned short, fh) << (16 * h);
      lw |= (unsigned)__builtin_bit_cast(unsigned short, fl) << (16 * h);
    }
    kfH[i] = hw; kfL[i] = lw;
  }
  // x slab with t-halo (SAME padding -> zeros outside [0,T)), split hi/lo f16.
  for (int i = tid; i < (TPB + 2) * NP * DP; i += THREADS) {
    int it = i / (NP * DP);
    int pd = i % (NP * DP);
    int tg = t0 - 1 + it;
    float f = (tg >= 0 && tg < T_SZ) ? x[(size_t)(b * T_SZ + tg) * (NP * DP) + pd] : 0.0f;
    _Float16 fh = (_Float16)f;
    xsH[i] = __builtin_bit_cast(unsigned short, fh);
    xsL[i] = __builtin_bit_cast(unsigned short, (_Float16)(f - (float)fh));
  }
  // B_weight slab (contiguous copy).
  for (int i = tid; i < TPB * NT * NP; i += THREADS)
    bws[i] = bw[(size_t)t0 * NT * NP + i];
  __syncthreads();

  const float RS8 = 0.35355339059327373f; // 1/sqrt(8)

  // ================= Per-wave t loop =================
  for (int tt = 0; tt < T_PER_WAVE; ++tt) {
    const int tl = wave * T_PER_WAVE + tt;
    const int t  = t0 + tl;

    // ---- Build X A-fragments (2 p-tiles, hi & lo), bias row at k==24 ----
    v16h xah[2], xal[2];
#pragma unroll
    for (int pt = 0; pt < 2; ++pt) {
      int p = pt * 16 + col;
#pragma unroll
      for (int r = 0; r < 8; ++r) {
        int kb = 2 * (r & 3) + ((r >= 4) ? 16 : 0) + (hiL ? 8 : 0);
#pragma unroll
        for (int h = 0; h < 2; ++h) {
          int k = kb + h;
          _Float16 vh = (_Float16)0.0f, vl = (_Float16)0.0f;
          if (k < 24) {
            int xi = (tl + (k >> 3)) * (NP * DP) + p * DP + (k & 7);
            vh = __builtin_bit_cast(_Float16, xsH[xi]);
            vl = __builtin_bit_cast(_Float16, xsL[xi]);
          } else if (k == 24) {
            vh = (_Float16)1.0f;   // bias row
          }
          xah[pt][2 * r + h] = vh;
          xal[pt][2 * r + h] = vl;
        }
      }
    }

    // ---- Conv GEMM: Y = X*K (split-f16, 3 products), relu, v, stage Y ----
    for (int m = 0; m < 16; ++m) {
      FragU KH, KL;
#pragma unroll
      for (int r = 0; r < 8; ++r) {
        KH.u[r] = kfH[(m * 32 + lane) * 8 + r];
        KL.u[r] = kfL[(m * 32 + lane) * 8 + r];
      }
      v8f a0 = {}; v8f a1 = {};
      a0 = __builtin_amdgcn_wmma_f32_16x16x32_f16(false, xah[0], false, KH.v, (short)0, a0, false, false);
      a0 = __builtin_amdgcn_wmma_f32_16x16x32_f16(false, xal[0], false, KH.v, (short)0, a0, false, false);
      a0 = __builtin_amdgcn_wmma_f32_16x16x32_f16(false, xah[0], false, KL.v, (short)0, a0, false, false);
      a1 = __builtin_amdgcn_wmma_f32_16x16x32_f16(false, xah[1], false, KH.v, (short)0, a1, false, false);
      a1 = __builtin_amdgcn_wmma_f32_16x16x32_f16(false, xal[1], false, KH.v, (short)0, a1, false, false);
      a1 = __builtin_amdgcn_wmma_f32_16x16x32_f16(false, xah[1], false, KL.v, (short)0, a1, false, false);

      float vs = 0.0f;
#pragma unroll
      for (int r = 0; r < 8; ++r) {
        a0[r] = fmaxf(a0[r], 0.0f);
        a1[r] = fmaxf(a1[r], 0.0f);
        vs += a0[r] + a1[r];
      }
      vs += __shfl_xor(vs, 16, 32);                  // sum both lane halves -> v[o]
      if (!hiL) vbuf[wave][m * 16 + col] = __builtin_bit_cast(unsigned short, (_Float16)vs);

      const int o  = m * 16 + col;
      const int pb = hiL ? 4 : 0;
#pragma unroll
      for (int r = 0; r < 8; r += 2) {               // D-regs are p-rows: pack pairs along p
        ybuf[wave][o * 16 + pb + (r >> 1)]     = pk2(a0[r], a0[r + 1]);
        ybuf[wave][o * 16 + pb + 8 + (r >> 1)] = pk2(a1[r], a1[r + 1]);
      }
    }

    // ---- E phase: s = U*v per head (2 heads per WMMA), softmax, w = c + Bw ----
#pragma unroll
    for (int j = 0; j < 8; ++j) {
      const int n0 = 2 * j, n1 = 2 * j + 1;
      // B fragment: column-gated broadcast of v (cols 0..7 -> n0, 8..15 -> n1)
      v16h vb;
#pragma unroll
      for (int r = 0; r < 8; ++r) {
        int kb = 2 * (r & 3) + ((r >= 4) ? 16 : 0) + (hiL ? 8 : 0);
#pragma unroll
        for (int h = 0; h < 2; ++h) {
          int k = kb + h;
          int head = (k < 16) ? n0 : n1;
          bool act = ((col < 8) == (k < 16));
          _Float16 f = (_Float16)0.0f;
          if (act) f = __builtin_bit_cast(_Float16, vbuf[wave][head * 16 + (k & 15)]);
          vb[2 * r + h] = f;
        }
      }
      // A fragments: U-pair (transposed gather from ybuf), one WMMA per q-tile
      v8f e[2];
#pragma unroll
      for (int qt = 0; qt < 2; ++qt) {
        const int q = qt * 16 + col;
        v16h ua;
#pragma unroll
        for (int r = 0; r < 8; ++r) {
          int kb = 2 * (r & 3) + ((r >= 4) ? 16 : 0) + (hiL ? 8 : 0);
#pragma unroll
          for (int h = 0; h < 2; ++h) {
            int k = kb + h;
            int head = (k < 16) ? n0 : n1;
            int idx16 = (head * 16 + (k & 15)) * 32 + q;   // o*32 + p
            ua[2 * r + h] = h16(ybuf[wave][idx16 >> 1], idx16 & 1);
          }
        }
        v8f z = {};
        e[qt] = __builtin_amdgcn_wmma_f32_16x16x32_f16(false, ua, false, vb, (short)0, z, false, false);
      }
      // Softmax over q=0..31 (this lane: 16 q's; partner lane L^16 holds the rest)
      float sv[16];
#pragma unroll
      for (int r = 0; r < 8; ++r) { sv[r] = e[0][r] * RS8; sv[8 + r] = e[1][r] * RS8; }
      float mx = sv[0];
#pragma unroll
      for (int i2 = 1; i2 < 16; ++i2) mx = fmaxf(mx, sv[i2]);
      mx = fmaxf(mx, __shfl_xor(mx, 16, 32));
      float ex[16]; float sum = 0.0f;
#pragma unroll
      for (int i2 = 0; i2 < 16; ++i2) { ex[i2] = __expf(sv[i2] - mx); sum += ex[i2]; }
      sum += __shfl_xor(sum, 16, 32);
      float inv = __builtin_amdgcn_rcpf(sum);
      // Each lane is the unique writer of 2 entries of w[n][q]
      const int nh = (col < 8) ? n0 : n1;
      const int r0 = col & 7;
      const int q0 = r0 + (hiL ? 8 : 0);
      float c0 = ex[r0]     * inv + bws[(tl * NT + nh) * NP + q0];
      float c1 = ex[8 + r0] * inv + bws[(tl * NT + nh) * NP + q0 + 16];
      wbuf[wave][nh * NP + q0]      = __builtin_bit_cast(unsigned short, (_Float16)c0);
      wbuf[wave][nh * NP + q0 + 16] = __builtin_bit_cast(unsigned short, (_Float16)c1);
    }

    // ---- O phase: out_n = w_n^T * U_n via block-diag of W*Y, then squash-norm ----
    v16h wa;
#pragma unroll
    for (int r = 0; r < 8; ++r) {
      int kb = 2 * (r & 3) + ((r >= 4) ? 16 : 0) + (hiL ? 8 : 0);
#pragma unroll
      for (int h = 0; h < 2; ++h)
        wa[2 * r + h] = __builtin_bit_cast(_Float16, wbuf[wave][col * NP + kb + h]);
    }
#pragma unroll
    for (int u = 0; u < 16; ++u) {
      FragU yb;
#pragma unroll
      for (int r = 0; r < 8; ++r) {
        int pp = (r & 3) + ((r >= 4) ? 8 : 0) + (hiL ? 4 : 0);
        yb.u[r] = ybuf[wave][(u * 16 + col) * 16 + pp];
      }
      v8f z = {};
      v8f od = __builtin_amdgcn_wmma_f32_16x16x32_f16(false, wa, false, yb.v, (short)0, z, false, false);
      float val = od[u & 7];            // row n == u lives in one lane-half
      float ss = val * val;             // reduce over d (16 lanes of the half)
      ss += __shfl_xor(ss, 1, 32);
      ss += __shfl_xor(ss, 2, 32);
      ss += __shfl_xor(ss, 4, 32);
      ss += __shfl_xor(ss, 8, 32);
      if (hiL == (u >= 8)) {
        float nrm   = __builtin_amdgcn_sqrtf(ss);
        float scale = ss / ((ss + 1.0f) * (nrm + 1e-7f));
        out[(((size_t)(b * T_SZ + t)) * NT + u) * DT + col] = val * scale;
      }
    }
  }
}

extern "C" void kernel_launch(void* const* d_in, const int* in_sizes, int n_in,
                              void* d_out, int out_size, void* d_ws, size_t ws_size,
                              hipStream_t stream) {
  (void)in_sizes; (void)n_in; (void)d_ws; (void)ws_size; (void)out_size;
  const float* x  = (const float*)d_in[0];
  const float* ck = (const float*)d_in[1];
  const float* cb = (const float*)d_in[2];
  const float* bw = (const float*)d_in[3];
  float* o = (float*)d_out;
  caps_wmma_kernel<<<dim3(B_SZ * (T_SZ / TPB)), dim3(THREADS), 0, stream>>>(x, ck, cb, bw, o);
}
